// TransformerBlock_56822417326259
// MI455X (gfx1250) — compile-verified, matrix operand order checked by reading, not run on hardware
//
#include <hip/hip_runtime.h>
#include <math.h>

// ---------------------------------------------------------------------------
// MI455X (gfx1250) transformer block.
//  - All GEMMs on v_wmma_f32_16x16x32_bf16 (bf16 A/B, fp32 accumulate).
//  - Wave32: each wave owns a 64x64 C tile = 4x4 grid of 16x16 WMMA tiles.
//  - Every GEMM uses the NT path (both operands row-major over K) with
//    contiguous 128-bit fragment loads; v is pre-transposed per head so
//    attn@v is NT too.
//  - K-loop is branch-free (row indices clamped; garbage rows/cols are
//    discarded by the bounds-checked store) and ping-pong double-buffered so
//    global loads overlap the WMMA pipe.
// ---------------------------------------------------------------------------

typedef __attribute__((ext_vector_type(16))) __bf16 v16bf;
typedef __attribute__((ext_vector_type(8)))  float  v8f;
typedef __attribute__((ext_vector_type(4)))  unsigned int u32x4;
typedef unsigned short bfraw;   // raw bf16 storage

#define DIMN   768
#define NTOK   1024
#define BATCH  4
#define NHEAD  12
#define HDIM   64
#define GPN    49
#define MLPN   3072
#define ROWS   (BATCH * NTOK)        // 4096
#define ZBH    (BATCH * NHEAD)       // 48

// ---------------- helpers ----------------

__device__ __forceinline__ bfraw f2bf(float x) {
  unsigned u = __float_as_uint(x);
  u += 0x7fffu + ((u >> 16) & 1u);          // round-to-nearest-even
  return (bfraw)(u >> 16);
}

__device__ __forceinline__ float gelu_exact(float x) {
  return 0.5f * x * (1.0f + erff(x * 0.70710678118654752f));
}

union FragU { u32x4 q[2]; v16bf v; bfraw u[16]; };

// 16x32 bf16 fragment, row-major source, lane L: row = L&15, K-half base
// (L>>4)*8 already folded into the pointer. Two contiguous 16-byte loads.
__device__ __forceinline__ v16bf load_pair(const bfraw* __restrict__ p) {
  FragU f;
  f.q[0] = *reinterpret_cast<const u32x4*>(p);
  f.q[1] = *reinterpret_cast<const u32x4*>(p + 16);
  return f.v;
}

__device__ __forceinline__ void load_set(const bfraw* const ap[4],
                                         const bfraw* const bp[4], int kb,
                                         v16bf a[4], v16bf b[4]) {
#pragma unroll
  for (int i = 0; i < 4; ++i) a[i] = load_pair(ap[i] + kb);
#pragma unroll
  for (int j = 0; j < 4; ++j) b[j] = load_pair(bp[j] + kb);
}

__device__ __forceinline__ void mma_set(const v16bf (&a)[4], const v16bf (&b)[4],
                                        v8f (&acc)[4][4]) {
#pragma unroll
  for (int i = 0; i < 4; ++i)
#pragma unroll
    for (int j = 0; j < 4; ++j)
      acc[i][j] = __builtin_amdgcn_wmma_f32_16x16x32_bf16(
          false, a[i], false, b[j], (short)0, acc[i][j], false, false);
}

// ---------------- batched WMMA GEMM (NT) ----------------
// C[M,N] = A[M,K] (bf16, row-major) * B[N,K]^T (bf16, row-major over K)
// [+bias][gelu][+resid], out fp32 or bf16.  Requires K % 64 == 0, K >= 64.
// Batch offsets: z -> (zb=z/zdiv, zh=z%zdiv); off = zb*s?b + zh*s?h.
template <bool BIAS, bool GELU_, bool RES, bool OBF16>
__global__ __launch_bounds__(256) void gemm_wmma_kernel(
    const bfraw* __restrict__ A, const bfraw* __restrict__ B,
    const float* __restrict__ bias, const float* __restrict__ resid,
    void* __restrict__ Cv,
    int M, int N, int K, int lda, int ldb, int ldc, int zdiv,
    long long sAb, long long sAh, long long sBb, long long sBh,
    long long sCb, long long sCh, long long sR) {
  const int z  = blockIdx.z;
  const int zb = z / zdiv, zh = z % zdiv;
  const bfraw* Az = A + (long long)zb * sAb + (long long)zh * sAh;
  const bfraw* Bz = B + (long long)zb * sBb + (long long)zh * sBh;
  const long long cOff = (long long)zb * sCb + (long long)zh * sCh;

  const int lane  = threadIdx.x & 31;
  const int wave  = threadIdx.x >> 5;
  const int tilesN = (N + 63) >> 6;
  const int tilesM = (M + 63) >> 6;
  const long long tid = (long long)blockIdx.x * 8 + wave;
  if (tid >= (long long)tilesM * tilesN) return;
  const int tm = (int)(tid / tilesN);
  const int tn = (int)(tid - (long long)tm * tilesN);
  const int row0 = tm << 6, col0 = tn << 6;
  const int lrow  = lane & 15;
  const int khalf = (lane >> 4) << 3;

  // Branch-free fragment pointers: clamp to last valid row; out-of-range
  // rows/cols only pollute C entries that the store masks off anyway.
  const bfraw* ap[4];
  const bfraw* bp[4];
#pragma unroll
  for (int i = 0; i < 4; ++i) {
    int ar = row0 + i * 16 + lrow; ar = (ar < M) ? ar : (M - 1);
    int bc = col0 + i * 16 + lrow; bc = (bc < N) ? bc : (N - 1);
    ap[i] = Az + (long long)ar * lda + khalf;
    bp[i] = Bz + (long long)bc * ldb + khalf;
  }

  v8f acc[4][4];
  const v8f vz = {0.f, 0.f, 0.f, 0.f, 0.f, 0.f, 0.f, 0.f};
#pragma unroll
  for (int i = 0; i < 4; ++i)
#pragma unroll
    for (int j = 0; j < 4; ++j) acc[i][j] = vz;

  // Ping-pong double-buffered K loop (two 32-wide steps per iteration).
  v16bf a0[4], b0[4], a1[4], b1[4];
  load_set(ap, bp, 0, a0, b0);
  for (int kb = 0; kb < K; kb += 64) {
    load_set(ap, bp, kb + 32, a1, b1);
    mma_set(a0, b0, acc);
    if (kb + 64 < K) load_set(ap, bp, kb + 64, a0, b0);
    mma_set(a1, b1, acc);
  }

  // C/D layout: lane = col within half, VGPR r = row (rows 8..15 for lanes>=16)
  const int ncl = lane & 15;
  const int mof = (lane >> 4) << 3;
#pragma unroll
  for (int j = 0; j < 4; ++j) {
    const int ng = col0 + j * 16 + ncl;
    if (ng >= N) continue;
    const float bv = BIAS ? bias[ng] : 0.f;
#pragma unroll
    for (int i = 0; i < 4; ++i) {
      const int mbase = row0 + i * 16 + mof;
#pragma unroll
      for (int r = 0; r < 8; ++r) {
        const int mg = mbase + r;
        if (mg >= M) continue;
        float vv = acc[i][j][r] + bv;
        if (GELU_) vv = gelu_exact(vv);
        if (RES)   vv += resid[(long long)z * sR + (long long)mg * ldc + ng];
        const long long ci = cOff + (long long)mg * ldc + ng;
        if (OBF16) ((bfraw*)Cv)[ci] = f2bf(vv);
        else       ((float*)Cv)[ci] = vv;
      }
    }
  }
}

// ---------------- LayerNorm (row=768) -> bf16 ----------------
__global__ __launch_bounds__(256) void ln_bf16_kernel(
    const float* __restrict__ x, const float* __restrict__ w,
    const float* __restrict__ b, bfraw* __restrict__ out) {
  const int row = blockIdx.x;
  const float* xr = x + (long long)row * DIMN;
  const int t = threadIdx.x;
  float v0 = xr[t], v1 = xr[t + 256], v2 = xr[t + 512];
  __shared__ float sred[256];
  sred[t] = v0 + v1 + v2;
  __syncthreads();
  for (int off = 128; off > 0; off >>= 1) {
    if (t < off) sred[t] += sred[t + off];
    __syncthreads();
  }
  const float mu = sred[0] * (1.f / (float)DIMN);
  __syncthreads();
  const float d0 = v0 - mu, d1 = v1 - mu, d2 = v2 - mu;
  sred[t] = d0 * d0 + d1 * d1 + d2 * d2;
  __syncthreads();
  for (int off = 128; off > 0; off >>= 1) {
    if (t < off) sred[t] += sred[t + off];
    __syncthreads();
  }
  const float rs = rsqrtf(sred[0] * (1.f / (float)DIMN) + 1e-5f);
  bfraw* orow = out + (long long)row * DIMN;
  orow[t]       = f2bf(d0 * rs * w[t]       + b[t]);
  orow[t + 256] = f2bf(d1 * rs * w[t + 256] + b[t + 256]);
  orow[t + 512] = f2bf(d2 * rs * w[t + 512] + b[t + 512]);
}

// ---------------- weight convert/transpose: fp32 [K,N] -> bf16 [N,K] --------
__global__ void convert_wt_kernel(const float* __restrict__ w,
                                  bfraw* __restrict__ wt, int K, int N) {
  long long idx = (long long)blockIdx.x * 256 + threadIdx.x;
  if (idx >= (long long)K * N) return;
  int n = (int)(idx / K);
  int k = (int)(idx - (long long)n * K);
  wt[idx] = f2bf(w[(long long)k * N + n]);
}

__global__ void convert_bf_kernel(const float* __restrict__ w,
                                  bfraw* __restrict__ o, int n) {
  int idx = blockIdx.x * 256 + threadIdx.x;
  if (idx < n) o[idx] = f2bf(w[idx]);
}

// ---------------- per-head transpose of v: [b,m,h*64+d] -> [z,d,m] ----------
__global__ void vtrans_kernel(const bfraw* __restrict__ qkv,
                              bfraw* __restrict__ vt) {
  const long long idx = (long long)blockIdx.x * 256 + threadIdx.x; // 48*64*1024
  const int m = (int)(idx & (NTOK - 1));
  const int d = (int)((idx >> 10) & (HDIM - 1));
  const int z = (int)(idx >> 16);
  const int b = z / NHEAD, h = z % NHEAD;
  vt[idx] = qkv[((long long)(b * NTOK + m)) * (3 * DIMN) + 2 * DIMN + h * HDIM + d];
}

// ---------------- gelu + softmax over 49 groups (one wave per row) ----------
__global__ __launch_bounds__(128) void gp_softmax_kernel(
    const float* __restrict__ graw, bfraw* __restrict__ gsm) {
  const int lane = threadIdx.x & 31;
  const int wv   = threadIdx.x >> 5;
  const long long row = (long long)blockIdx.x * 4 + wv;
  const float* in = graw + row * 64;
  float e0 = (lane < GPN)      ? gelu_exact(in[lane])      : -3.4e38f;
  float e1 = (lane + 32 < GPN) ? gelu_exact(in[lane + 32]) : -3.4e38f;
  float mx = fmaxf(e0, e1);
#pragma unroll
  for (int off = 16; off > 0; off >>= 1) mx = fmaxf(mx, __shfl_xor(mx, off, 32));
  float p0 = (lane < GPN)      ? expf(e0 - mx) : 0.f;
  float p1 = (lane + 32 < GPN) ? expf(e1 - mx) : 0.f;
  float s = p0 + p1;
#pragma unroll
  for (int off = 16; off > 0; off >>= 1) s += __shfl_xor(s, off, 32);
  const float inv = 1.f / s;
  bfraw* o = gsm + row * 64;
  o[lane]      = f2bf(p0 * inv);       // cols >= 49 get exact zeros (K padding)
  o[lane + 32] = f2bf(p1 * inv);
}

// ---------------- attn = (1-a)*softmax(scores*scale*gw) + a*gw --------------
__global__ __launch_bounds__(256) void attn_mix_kernel(
    float* __restrict__ scores, const float* __restrict__ gwf,
    const float* __restrict__ alpha) {
  const long long rowi = blockIdx.x;            // 48*1024 rows
  const int z = (int)(rowi >> 10);
  const int h = z % NHEAD;
  const float a = 1.f / (1.f + expf(-alpha[h]));
  float* srow = scores + rowi * NTOK;
  const float* grow = gwf + rowi * NTOK;
  const int t = threadIdx.x;
  float gv[4], tv[4];
#pragma unroll
  for (int i = 0; i < 4; ++i) {
    const int c = t + 256 * i;
    gv[i] = grow[c];
    tv[i] = srow[c] * (0.125f * gv[i]);         // scale = 64^-0.5 folded in
  }
  __shared__ float sred[256];
  float mx = fmaxf(fmaxf(tv[0], tv[1]), fmaxf(tv[2], tv[3]));
  sred[t] = mx;
  __syncthreads();
  for (int off = 128; off > 0; off >>= 1) {
    if (t < off) sred[t] = fmaxf(sred[t], sred[t + off]);
    __syncthreads();
  }
  mx = sred[0];
  __syncthreads();
  float ev[4], s = 0.f;
#pragma unroll
  for (int i = 0; i < 4; ++i) { ev[i] = expf(tv[i] - mx); s += ev[i]; }
  sred[t] = s;
  __syncthreads();
  for (int off = 128; off > 0; off >>= 1) {
    if (t < off) sred[t] += sred[t + off];
    __syncthreads();
  }
  const float inv = 1.f / sred[0];
#pragma unroll
  for (int i = 0; i < 4; ++i) {
    const int c = t + 256 * i;
    srow[c] = (1.f - a) * (ev[i] * inv) + a * gv[i];
  }
}

// ---------------- column sums over queries (axis=2) -------------------------
__global__ void colsum_kernel(const float* __restrict__ attn,
                              float* __restrict__ colsum) {
  const int idx = blockIdx.x * 256 + threadIdx.x;   // 48*1024
  const int z = idx >> 10, m = idx & 1023;
  const float* p = attn + (long long)z * NTOK * NTOK + m;
  float s = 0.f;
  for (int n = 0; n < NTOK; ++n) s += p[(long long)n * NTOK];
  colsum[idx] = s;
}

// ---------------- column renorm + convert to bf16 ---------------------------
__global__ void renorm_bf16_kernel(const float* __restrict__ attn,
                                   const float* __restrict__ colsum,
                                   bfraw* __restrict__ out) {
  const long long idx = (long long)blockIdx.x * 256 + threadIdx.x;
  const int m = (int)(idx & 1023);
  const long long z = idx >> 20;                    // 1024*1024 per head
  const float cs = colsum[z * NTOK + m];
  out[idx] = f2bf(attn[idx] / (cs + 1e-8f));
}

// ---------------------------------------------------------------------------

extern "C" void kernel_launch(void* const* d_in, const int* in_sizes, int n_in,
                              void* d_out, int out_size, void* d_ws, size_t ws_size,
                              hipStream_t stream) {
  const float* x      = (const float*)d_in[0];
  const float* ln1_w  = (const float*)d_in[1];
  const float* ln1_b  = (const float*)d_in[2];
  const float* qkv_w  = (const float*)d_in[3];
  const float* qkv_b  = (const float*)d_in[4];
  const float* proj_w = (const float*)d_in[5];
  const float* proj_b = (const float*)d_in[6];
  const float* gp_w   = (const float*)d_in[7];
  const float* alpha  = (const float*)d_in[8];
  const float* ln2_w  = (const float*)d_in[9];
  const float* ln2_b  = (const float*)d_in[10];
  const float* ff1_w  = (const float*)d_in[11];
  const float* ff1_b  = (const float*)d_in[12];
  const float* ff2_w  = (const float*)d_in[13];
  const float* ff2_b  = (const float*)d_in[14];
  float* out = (float*)d_out;
  (void)in_sizes; (void)n_in; (void)out_size; (void)ws_size;

  // ---- workspace arena (~620 MB) ----
  char* p = (char*)d_ws;
  auto carve = [&](size_t bytes) -> void* {
    void* r = (void*)p;
    p += (bytes + 255) & ~((size_t)255);
    return r;
  };
  bfraw* qkv_wt   = (bfraw*)carve((size_t)2304 * 768 * 2);       // [N,K]
  bfraw* proj_wt  = (bfraw*)carve((size_t)768  * 768 * 2);
  bfraw* ff1_wt   = (bfraw*)carve((size_t)3072 * 768 * 2);
  bfraw* ff2_wt   = (bfraw*)carve((size_t)768  * 3072 * 2);
  bfraw* gp_bf    = (bfraw*)carve((size_t)NHEAD * GPN * HDIM * 2);
  bfraw* xn_bf    = (bfraw*)carve((size_t)ROWS * DIMN * 2);
  bfraw* qkv_bf   = (bfraw*)carve((size_t)ROWS * 3 * DIMN * 2);
  bfraw* vt_bf    = (bfraw*)carve((size_t)ZBH * HDIM * NTOK * 2);
  float* gwraw    = (float*)carve((size_t)ZBH * NTOK * 64 * 4);
  bfraw* gwsm_bf  = (bfraw*)carve((size_t)ZBH * NTOK * 64 * 2);
  float* scores   = (float*)carve((size_t)ZBH * NTOK * NTOK * 4);
  float* gwf      = (float*)carve((size_t)ZBH * NTOK * NTOK * 4);
  float* colsum   = (float*)carve((size_t)ZBH * NTOK * 4);
  bfraw* attn_bf  = (bfraw*)carve((size_t)ZBH * NTOK * NTOK * 2);
  bfraw* oattn_bf = (bfraw*)carve((size_t)ROWS * DIMN * 2);
  float* x1       = (float*)carve((size_t)ROWS * DIMN * 4);
  bfraw* xn2_bf   = (bfraw*)carve((size_t)ROWS * DIMN * 2);
  bfraw* hmid_bf  = (bfraw*)carve((size_t)ROWS * MLPN * 2);

  auto ggrid = [](int M, int N, int Z) {
    long long tiles = (long long)((M + 63) / 64) * ((N + 63) / 64);
    return dim3((unsigned)((tiles + 7) / 8), 1, (unsigned)Z);
  };

  // ---- weights -> bf16 (transposed to [N,K]) ----
  convert_wt_kernel<<<(768 * 2304 + 255) / 256, 256, 0, stream>>>(qkv_w,  qkv_wt,  768,  2304);
  convert_wt_kernel<<<(768 * 768  + 255) / 256, 256, 0, stream>>>(proj_w, proj_wt, 768,  768);
  convert_wt_kernel<<<(768 * 3072 + 255) / 256, 256, 0, stream>>>(ff1_w,  ff1_wt,  768,  3072);
  convert_wt_kernel<<<(3072 * 768 + 255) / 256, 256, 0, stream>>>(ff2_w,  ff2_wt,  3072, 768);
  convert_bf_kernel<<<(NHEAD * GPN * HDIM + 255) / 256, 256, 0, stream>>>(gp_w, gp_bf, NHEAD * GPN * HDIM);

  // ---- LN1 ----
  ln_bf16_kernel<<<ROWS, 256, 0, stream>>>(x, ln1_w, ln1_b, xn_bf);

  // ---- QKV: [4096,768] x [768,2304] + b -> bf16 ----
  gemm_wmma_kernel<true, false, false, true>
      <<<ggrid(ROWS, 3 * DIMN, 1), 256, 0, stream>>>(
          xn_bf, qkv_wt, qkv_b, nullptr, qkv_bf,
          ROWS, 3 * DIMN, DIMN, DIMN, DIMN, 3 * DIMN, 1,
          0, 0, 0, 0, 0, 0, 0);

  // ---- v transpose per head: [z][d][m] ----
  vtrans_kernel<<<(ZBH * HDIM * NTOK) / 256, 256, 0, stream>>>(qkv_bf, vt_bf);

  // ---- scores = q @ k^T (per b,h): A=q rows, B(NT)=k rows ----
  gemm_wmma_kernel<false, false, false, false>
      <<<ggrid(NTOK, NTOK, ZBH), 256, 0, stream>>>(
          qkv_bf, qkv_bf + DIMN, nullptr, nullptr, scores,
          NTOK, NTOK, HDIM, 3 * DIMN, 3 * DIMN, NTOK, NHEAD,
          (long long)NTOK * 3 * DIMN, HDIM,
          (long long)NTOK * 3 * DIMN, HDIM,
          (long long)NHEAD * NTOK * NTOK, (long long)NTOK * NTOK, 0);

  // ---- gw_raw = v @ gp[h]^T : [1024,64] x [49,64]^T ----
  gemm_wmma_kernel<false, false, false, false>
      <<<ggrid(NTOK, GPN, ZBH), 256, 0, stream>>>(
          qkv_bf + 2 * DIMN, gp_bf, nullptr, nullptr, gwraw,
          NTOK, GPN, HDIM, 3 * DIMN, HDIM, 64, NHEAD,
          (long long)NTOK * 3 * DIMN, HDIM,
          0, (long long)GPN * HDIM,
          (long long)NHEAD * NTOK * 64, (long long)NTOK * 64, 0);

  // ---- gelu + softmax over 49 -> bf16 (K padded to 64 with zeros) ----
  gp_softmax_kernel<<<(ZBH * NTOK) / 4, 128, 0, stream>>>(gwraw, gwsm_bf);

  // ---- gw_full = gwsm @ gwsm^T : [1024,64] x [1024,64]^T ----
  gemm_wmma_kernel<false, false, false, false>
      <<<ggrid(NTOK, NTOK, ZBH), 256, 0, stream>>>(
          gwsm_bf, gwsm_bf, nullptr, nullptr, gwf,
          NTOK, NTOK, 64, 64, 64, NTOK, 1,
          (long long)NTOK * 64, 0, (long long)NTOK * 64, 0,
          (long long)NTOK * NTOK, 0, 0);

  // ---- softmax(scores*scale*gw) mixed with gw (in place over scores) ----
  attn_mix_kernel<<<ZBH * NTOK, 256, 0, stream>>>(scores, gwf, alpha);

  // ---- column renorm (axis=2) ----
  colsum_kernel<<<(ZBH * NTOK) / 256, 256, 0, stream>>>(scores, colsum);
  renorm_bf16_kernel<<<(unsigned)(((long long)ZBH * NTOK * NTOK) / 256), 256, 0, stream>>>(
      scores, colsum, attn_bf);

  // ---- out = attn @ v : A=attn rows, B(NT)=v^T rows (d-major) ----
  gemm_wmma_kernel<false, false, false, true>
      <<<ggrid(NTOK, HDIM, ZBH), 256, 0, stream>>>(
          attn_bf, vt_bf, nullptr, nullptr, oattn_bf,
          NTOK, HDIM, NTOK, NTOK, NTOK, DIMN, NHEAD,
          (long long)NHEAD * NTOK * NTOK, (long long)NTOK * NTOK,
          (long long)NHEAD * HDIM * NTOK, (long long)HDIM * NTOK,
          (long long)NTOK * DIMN, HDIM, 0);

  // ---- proj + bias + residual(x) -> x1 fp32 ----
  gemm_wmma_kernel<true, false, true, false>
      <<<ggrid(ROWS, DIMN, 1), 256, 0, stream>>>(
          oattn_bf, proj_wt, proj_b, x, x1,
          ROWS, DIMN, DIMN, DIMN, DIMN, DIMN, 1,
          0, 0, 0, 0, 0, 0, 0);

  // ---- LN2 ----
  ln_bf16_kernel<<<ROWS, 256, 0, stream>>>(x1, ln2_w, ln2_b, xn2_bf);

  // ---- FF1 + bias + gelu -> bf16 ----
  gemm_wmma_kernel<true, true, false, true>
      <<<ggrid(ROWS, MLPN, 1), 256, 0, stream>>>(
          xn2_bf, ff1_wt, ff1_b, nullptr, hmid_bf,
          ROWS, MLPN, DIMN, DIMN, DIMN, MLPN, 1,
          0, 0, 0, 0, 0, 0, 0);

  // ---- FF2 + bias + residual(x1) -> d_out fp32 ----
  gemm_wmma_kernel<true, false, true, false>
      <<<ggrid(ROWS, DIMN, 1), 256, 0, stream>>>(
          hmid_bf, ff2_wt, ff2_b, x1, out,
          ROWS, DIMN, MLPN, MLPN, MLPN, DIMN, 1,
          0, 0, 0, 0, 0, 0, 0);
}
